// Attention_44564580663760
// MI455X (gfx1250) — matrix operand order repaired and checked by reading, not compiled, not run on hardware
//
#include <hip/hip_runtime.h>
#include <hip/hip_bf16.h>
#include <stdint.h>
#include <stddef.h>

// ---------------------------------------------------------------------------
// Problem constants (B=8, S=2048, D=1024)
// ---------------------------------------------------------------------------
static constexpr int kB = 8;
static constexpr int kS = 2048;
static constexpr int kD = 1024;
static constexpr int kM = kB * kS;      // 16384 flattened rows

// Tiling: 128x128 block, K-step 32, 8 waves (256 thr), 32x64 per wave
static constexpr int BM = 128, BN = 128, BK = 32;
static constexpr int LDT = BK + 2;      // padded LDS stride (shorts)

typedef __attribute__((ext_vector_type(16))) __bf16   v16bf;
typedef __attribute__((ext_vector_type(8)))  float    v8f;
typedef __attribute__((ext_vector_type(8)))  unsigned v8u;
typedef __attribute__((ext_vector_type(4)))  unsigned uint32x4;
typedef __attribute__((ext_vector_type(8)))  int      int32x8;
typedef __attribute__((ext_vector_type(4)))  int      int32x4;

#if __has_builtin(__builtin_amdgcn_tensor_load_to_lds) && \
    __has_builtin(__builtin_amdgcn_s_wait_tensorcnt)
#define USE_TDM 1
#else
#define USE_TDM 0
#endif

// ---------------------------------------------------------------------------
// bf16 helpers (round-to-nearest-even)
// ---------------------------------------------------------------------------
__device__ __forceinline__ unsigned short f2bf_bits(float f) {
  unsigned u = __builtin_bit_cast(unsigned, f);
  unsigned r = u + 0x7fffu + ((u >> 16) & 1u);
  return (unsigned short)(r >> 16);
}
__device__ __forceinline__ float bf_bits2f(unsigned short h) {
  unsigned u = ((unsigned)h) << 16;
  return __builtin_bit_cast(float, u);
}

__device__ __forceinline__ v8f wmma_bf16(v16bf a, v16bf b, v8f c) {
  return __builtin_amdgcn_wmma_f32_16x16x32_bf16(
      false, a, false, b, (short)0, c, false, false);
}

// ---------------------------------------------------------------------------
// WMMA fragment loads from padded LDS tiles (layouts per CDNA5 ISA)
// ---------------------------------------------------------------------------
__device__ __forceinline__ v16bf frag_A(const unsigned short* t, int lane) {
  const int m  = lane & 15;
  const int kb = (lane >> 4) << 3;            // 0 / 8
  const unsigned short* p = t + m * LDT;
  v8u r;
#pragma unroll
  for (int i = 0; i < 4; ++i) r[i]     = *(const unsigned*)(p + kb + 2 * i);
#pragma unroll
  for (int i = 0; i < 4; ++i) r[4 + i] = *(const unsigned*)(p + 16 + kb + 2 * i);
  return __builtin_bit_cast(v16bf, r);
}

__device__ __forceinline__ v16bf frag_B(const unsigned short* t, int lane) {
  const int n  = lane & 15;
  const int kb = (lane >> 4) << 3;            // 0 / 8
  const unsigned short* p = t + n * LDT;
  v8u r;
#pragma unroll
  for (int i = 0; i < 4; ++i) r[i]     = *(const unsigned*)(p + kb + 2 * i);
#pragma unroll
  for (int i = 0; i < 4; ++i) r[4 + i] = *(const unsigned*)(p + 16 + kb + 2 * i);
  return __builtin_bit_cast(v16bf, r);
}

// ---------------------------------------------------------------------------
// Tensor Data Mover: load a (rows x 32) bf16 tile (row stride ld_elems) into
// LDS with a 1-dword pad every 16 dwords -> LDS row stride = LDT shorts.
// D# packing per CDNA5 ISA. This toolchain's builtin is
// the 6-arg clang-23 form: (g0, g1, g2, g3, g4, cpol).
// ---------------------------------------------------------------------------
#if USE_TDM
__device__ __forceinline__ void tdm_tile_load(unsigned lds_byte_addr,
                                              const unsigned short* tile,
                                              unsigned rows, unsigned ld_elems) {
  const unsigned long long ga = (unsigned long long)(uintptr_t)tile;
  uint32x4 g0;
  g0[0] = 1u;                                     // count=1, user descriptor
  g0[1] = lds_byte_addr;                          // lds_addr
  g0[2] = (unsigned)ga;                           // global_addr[31:0]
  g0[3] = (unsigned)((ga >> 32) & 0x01ffffffu) |  // global_addr[56:32]
          0x80000000u;                            // type=2 ("image")
  int32x8 g1;
  g1[0] = (int)((1u << 16) |                      // data_size = 2 bytes
                (1u << 20) |                      // pad_enable
                (3u << 22));                      // pad_interval = 16 dwords
                                                  // pad_amount  = 1 dword
  g1[1] = (int)((unsigned)BK << 16);              // tensor_dim0 = 32
  g1[2] = (int)((rows & 0xffffu) << 16);          // tensor_dim1 = rows
  g1[3] = (int)((unsigned)BK << 16);              // tile_dim0 = 32
  g1[4] = (int)rows;                              // tile_dim1 = rows
  g1[5] = (int)ld_elems;                          // tensor_dim0_stride (lo32)
  g1[6] = 0;
  g1[7] = 0;
  const int32x4 z4 = {0, 0, 0, 0};                // 2-D: groups 2/3 unused
  const int32x8 z8 = {0, 0, 0, 0, 0, 0, 0, 0};
  __builtin_amdgcn_tensor_load_to_lds(g0, g1, z4, z4, z8, 0);
}
#endif

// Low 32 bits of a generic pointer into LDS == wave-relative LDS byte offset
__device__ __forceinline__ unsigned lds_addr_of(const void* p) {
  return (unsigned)(uintptr_t)p;
}

// ---------------------------------------------------------------------------
// fp32 -> bf16 one-shot conversion
// ---------------------------------------------------------------------------
__global__ __launch_bounds__(256)
void cvt_f32_bf16(const float* __restrict__ s, unsigned short* __restrict__ d,
                  int n) {
  const int stride = gridDim.x * 256;
  for (int i = blockIdx.x * 256 + threadIdx.x; i < n; i += stride)
    d[i] = f2bf_bits(s[i]);
}

// ---------------------------------------------------------------------------
// Unified bf16 WMMA GEMM:  C[m,n] = scale * sum_k A[m,k]*B'[n,k] + bias[n]
//   OUT_F32 : write fp32 (else bf16 bits)
//   CN      : skip tiles fully above causal diagonal (scores)
//   CK      : prune K-loop to k < m0+BM (ctx, P==0 above diagonal)
//   BT      : B is row-major [K x N]; stage transposed (ctx V operand)
// Per-batch strides in elements; blockIdx.z = batch.
// ---------------------------------------------------------------------------
template <bool OUT_F32, bool CN, bool CK, bool BT>
__global__ __launch_bounds__(256)
void gemm_bf16_wmma(const unsigned short* __restrict__ A,
                    const unsigned short* __restrict__ Bm,
                    const float* __restrict__ bias, void* __restrict__ Cp,
                    int N, int K, long long sAb, long long sBb, long long sCb,
                    float scale) {
  const int b = blockIdx.z;
  A  += (size_t)b * sAb;
  Bm += (size_t)b * sBb;
  const int m0 = blockIdx.y * BM;
  const int n0 = blockIdx.x * BN;
  if (CN && n0 >= m0 + BM) return;                 // fully masked tile
  const int Kend = CK ? ((m0 + BM < K) ? m0 + BM : K) : K;

  __shared__ unsigned short As[2][BM][LDT];
  __shared__ unsigned short Bs[2][BN][LDT];

  const int tid = threadIdx.x, lane = tid & 31, wid = tid >> 5;
  const int wm = (wid >> 1) * 32;                  // 4 M-slices
  const int wn = (wid & 1) * 64;                   // 2 N-slices of 64

  v8f acc[2][4];
#pragma unroll
  for (int i = 0; i < 2; ++i)
#pragma unroll
    for (int j = 0; j < 4; ++j) acc[i][j] = (v8f){0,0,0,0,0,0,0,0};

  auto compute = [&](int cur) {
    const unsigned short (*pA)[LDT] = As[cur];
    const unsigned short (*pB)[LDT] = Bs[cur];
    v16bf a0 = frag_A(&pA[wm][0], lane);
    v16bf a1 = frag_A(&pA[wm + 16][0], lane);
#pragma unroll
    for (int j = 0; j < 4; ++j) {
      v16bf bj = frag_B(&pB[wn + 16 * j][0], lane);
      acc[0][j] = wmma_bf16(a0, bj, acc[0][j]);
      acc[1][j] = wmma_bf16(a1, bj, acc[1][j]);
    }
  };

  const int nt = Kend / BK;

#if USE_TDM
  if constexpr (!BT) {
    // Double-buffered: A+B tiles via Tensor Data Mover, wave 0 issues.
    const unsigned lA[2] = {lds_addr_of(&As[0][0][0]), lds_addr_of(&As[1][0][0])};
    const unsigned lB[2] = {lds_addr_of(&Bs[0][0][0]), lds_addr_of(&Bs[1][0][0])};
    if (wid == 0) {
      tdm_tile_load(lA[0], A + (size_t)m0 * K, BM, (unsigned)K);
      tdm_tile_load(lB[0], Bm + (size_t)n0 * K, BN, (unsigned)K);
    }
    for (int t = 0; t < nt; ++t) {
      const int cur = t & 1, nxt = cur ^ 1;
      if (wid == 0) {
        if (t + 1 < nt) {
          tdm_tile_load(lA[nxt], A + (size_t)m0 * K + (t + 1) * BK, BM, (unsigned)K);
          tdm_tile_load(lB[nxt], Bm + (size_t)n0 * K + (t + 1) * BK, BN, (unsigned)K);
          __builtin_amdgcn_s_wait_tensorcnt((short)2);  // tile t landed
        } else {
          __builtin_amdgcn_s_wait_tensorcnt((short)0);
        }
      }
      __syncthreads();          // publish tile t to all waves
      compute(cur);
      __syncthreads();          // done reading before buffer reuse
    }
  } else {
    // A via TDM (single buffer); B staged transposed cooperatively,
    // overlapped with the in-flight tensor DMA.
    const unsigned lA0 = lds_addr_of(&As[0][0][0]);
    for (int t = 0; t < nt; ++t) {
      if (wid == 0) tdm_tile_load(lA0, A + (size_t)m0 * K + t * BK, BM, (unsigned)K);
      for (int idx = tid; idx < BK * BN; idx += 256) {
        int r = idx >> 7, c = idx & (BN - 1);      // r = k, c = n
        Bs[0][c][r] = Bm[(size_t)(t * BK + r) * N + (n0 + c)];
      }
      if (wid == 0) __builtin_amdgcn_s_wait_tensorcnt((short)0);
      __syncthreads();
      compute(0);
      __syncthreads();
    }
  }
#else
  for (int t = 0; t < nt; ++t) {
    __syncthreads();
    for (int idx = tid; idx < BM * BK; idx += 256) {
      int r = idx >> 5, c = idx & 31;
      As[0][r][c] = A[(size_t)(m0 + r) * K + t * BK + c];
    }
    if constexpr (!BT) {
      for (int idx = tid; idx < BN * BK; idx += 256) {
        int r = idx >> 5, c = idx & 31;
        Bs[0][r][c] = Bm[(size_t)(n0 + r) * K + t * BK + c];
      }
    } else {
      for (int idx = tid; idx < BK * BN; idx += 256) {
        int r = idx >> 7, c = idx & (BN - 1);
        Bs[0][c][r] = Bm[(size_t)(t * BK + r) * N + (n0 + c)];
      }
    }
    if (t + 1 < nt)
      __builtin_prefetch(A + (size_t)(m0 + (tid >> 1)) * K + (t + 1) * BK, 0, 1);
    __syncthreads();
    compute(0);
  }
#endif

  // Epilogue: C/D layout -> row = vgpr + 8*(lane>=16), col = lane&15
  float*          Cf = (float*)Cp + (size_t)b * sCb;
  unsigned short* Ch = (unsigned short*)Cp + (size_t)b * sCb;
  const int cn = lane & 15, half = (lane >> 4) << 3;
#pragma unroll
  for (int j = 0; j < 4; ++j) {
    const int gcol = n0 + wn + 16 * j + cn;
    const float bval = bias ? bias[gcol] : 0.f;
#pragma unroll
    for (int i = 0; i < 2; ++i)
#pragma unroll
      for (int r = 0; r < 8; ++r) {
        const int grow = m0 + wm + 16 * i + half + r;
        const float v = acc[i][j][r] * scale + bval;
        if constexpr (OUT_F32) Cf[(size_t)grow * N + gcol] = v;
        else                   Ch[(size_t)grow * N + gcol] = f2bf_bits(v);
      }
  }
}

// ---------------------------------------------------------------------------
// One wave per row: causal softmax in-place over bf16 scores.
// ---------------------------------------------------------------------------
__global__ __launch_bounds__(256)
void softmax_rows(unsigned short* __restrict__ P) {
  const int wid = threadIdx.x >> 5, lane = threadIdx.x & 31;
  const int row = blockIdx.x * 8 + wid;            // over B*S rows
  const int m   = row & (kS - 1);
  unsigned short* rp = P + (size_t)row * kS;

  const int nvalid = m + 1;
  float vals[kS / 32];
  float mx = -3.0e38f;
#pragma unroll 8
  for (int i = 0; i < kS / 32; ++i) {
    const int c = lane + 32 * i;
    const float v = (c < nvalid) ? bf_bits2f(rp[c]) : -3.0e38f;
    vals[i] = v;
    mx = fmaxf(mx, v);
  }
#pragma unroll
  for (int off = 16; off; off >>= 1) mx = fmaxf(mx, __shfl_xor(mx, off, 32));

  float sum = 0.f;
#pragma unroll 8
  for (int i = 0; i < kS / 32; ++i) {
    const int c = lane + 32 * i;
    const float e = (c < nvalid) ? __expf(vals[i] - mx) : 0.f;
    vals[i] = e;
    sum += e;
  }
#pragma unroll
  for (int off = 16; off; off >>= 1) sum += __shfl_xor(sum, off, 32);

  const float inv = 1.0f / sum;
#pragma unroll 8
  for (int i = 0; i < kS / 32; ++i) {
    const int c = lane + 32 * i;
    rp[c] = f2bf_bits((c < nvalid) ? vals[i] * inv : 0.f);
  }
}

// ---------------------------------------------------------------------------
// Orchestration. Inputs: x, mask, Wq, bq, Wk, bk, Wv, bv, Wo, bo.
// Causal mask handled analytically; dead k-projection skipped.
// ---------------------------------------------------------------------------
extern "C" void kernel_launch(void* const* d_in, const int* in_sizes, int n_in,
                              void* d_out, int out_size, void* d_ws,
                              size_t ws_size, hipStream_t stream) {
  const float* x  = (const float*)d_in[0];
  const float* Wq = (const float*)d_in[2];
  const float* bq = (const float*)d_in[3];
  const float* Wv = (const float*)d_in[6];
  const float* bv = (const float*)d_in[7];
  const float* Wo = (const float*)d_in[8];
  const float* bo = (const float*)d_in[9];
  float* out = (float*)d_out;

  // Workspace (bf16 bits): xb | wq | wv | wo | q | v | P | ctx  (~200 MB)
  unsigned short* xb  = (unsigned short*)d_ws;
  unsigned short* wqb = xb  + (size_t)kM * kD;
  unsigned short* wvb = wqb + (size_t)kD * kD;
  unsigned short* wob = wvb + (size_t)kD * kD;
  unsigned short* qb  = wob + (size_t)kD * kD;
  unsigned short* vb  = qb  + (size_t)kM * kD;
  unsigned short* pb  = vb  + (size_t)kM * kD;
  unsigned short* cb  = pb  + (size_t)kB * kS * kS;

  const dim3 blk(256);

  // One-shot fp32 -> bf16 conversions (kills per-tile VALU conversion)
  cvt_f32_bf16<<<dim3(2048), blk, 0, stream>>>(x,  xb,  kM * kD);
  cvt_f32_bf16<<<dim3(512),  blk, 0, stream>>>(Wq, wqb, kD * kD);
  cvt_f32_bf16<<<dim3(512),  blk, 0, stream>>>(Wv, wvb, kD * kD);
  cvt_f32_bf16<<<dim3(512),  blk, 0, stream>>>(Wo, wob, kD * kD);

  const dim3 gproj(kD / BN, kM / BM, 1);   // (8, 128)
  const dim3 gsc(kS / BN, kS / BM, kB);    // (16, 16, 8)
  const dim3 gctx(kD / BN, kS / BM, kB);   // (8, 16, 8)

  // q / v projections
  gemm_bf16_wmma<false, false, false, false><<<gproj, blk, 0, stream>>>(
      xb, wqb, bq, qb, kD, kD, 0, 0, 0, 1.0f);
  gemm_bf16_wmma<false, false, false, false><<<gproj, blk, 0, stream>>>(
      xb, wvb, bv, vb, kD, kD, 0, 0, 0, 1.0f);
  // scores = q @ v^T / sqrt(D)  (reference's faithful bug), causal tiles only
  gemm_bf16_wmma<false, true, false, false><<<gsc, blk, 0, stream>>>(
      qb, vb, nullptr, pb, kS, kD,
      (long long)kS * kD, (long long)kS * kD, (long long)kS * kS, 0.03125f);
  softmax_rows<<<dim3(kB * kS / 8), blk, 0, stream>>>(pb);
  // ctx = P @ v  (K-loop causally pruned, V staged transposed)
  gemm_bf16_wmma<false, false, true, true><<<gctx, blk, 0, stream>>>(
      pb, vb, nullptr, cb, kD, kS,
      (long long)kS * kS, (long long)kS * kD, (long long)kS * kD, 1.0f);
  // out = ctx @ Wo^T + bo  (fp32)
  gemm_bf16_wmma<true, false, false, false><<<gproj, blk, 0, stream>>>(
      cb, wob, bo, out, kD, kD, 0, 0, 0, 1.0f);
}
